// MoeMLP_30107720745417
// MI455X (gfx1250) — matrix operand order repaired and checked by reading, not compiled, not run on hardware
//
#include <hip/hip_runtime.h>

// ---------------- problem constants ----------------
#define NT   8192          // tokens (4*2048)
#define ND   1024          // n_embd
#define NF   2048          // d_ffn per expert
#define NE   8             // experts
#define NK   2             // top-k
#define OUTN ((size_t)NT * ND)

typedef __attribute__((ext_vector_type(16))) __bf16 v16bf;
typedef __attribute__((ext_vector_type(8)))  float  v8f;
typedef __attribute__((ext_vector_type(4)))  unsigned int u32x4;
typedef __attribute__((ext_vector_type(8)))  int    i32x8;
typedef __attribute__((ext_vector_type(4)))  int    i32x4;

union Frag16 { v16bf v; u32x4 u[2]; };

#if defined(__gfx1250__) && __has_builtin(__builtin_amdgcn_tensor_load_to_lds) && \
    __has_builtin(__builtin_amdgcn_s_wait_tensorcnt) && (__clang_major__ < 23)
#define USE_TDM 1
#else
#define USE_TDM 0
#endif

// ---------------- workspace layout ----------------
struct WS {
  int   counts[8];     // header: first 34 words zeroed by init
  int   offsets[9];
  int   cursors[8];
  float z_acc;
  float p_sum[8];
  int   topi[NT * NK];
  float topw[NT * NK];
  int   row_token[NT * NK];
  float row_wt[NT * NK];
};
// bf16 scratch regions appended after WS (4 KB aligned)
#define XB_OFF   ((size_t)266240)                          // >= sizeof(WS)
#define XB_BYTES ((size_t)NT * ND * 2)                     // 16.8 MB
#define W1B_OFF  (XB_OFF + XB_BYTES)
#define W1B_BYTES ((size_t)ND * NE * NF * 2)               // 33.5 MB
#define W2B_OFF  (W1B_OFF + W1B_BYTES)

// ---------------- init: zero output + ws header ----------------
__global__ __launch_bounds__(256) void moe_init(float* __restrict__ out, WS* ws) {
  size_t idx = (size_t)blockIdx.x * 256 + threadIdx.x;
  if (idx < OUTN + 10) out[idx] = 0.0f;
  if (idx < 34) ((unsigned int*)ws)[idx] = 0u;
}

// ---------------- fp32 -> bf16 converts (one-time, memory-bound) ----------------
__global__ __launch_bounds__(256) void cvt_x(const float* __restrict__ x,
                                             __bf16* __restrict__ xb) {
  size_t i = (size_t)blockIdx.x * 256 + threadIdx.x;
  if (i < (size_t)NT * ND) xb[i] = (__bf16)x[i];
}

// w1 [D][E*F] -> fragment-native: dest[((e*128+ct)*32+kk)*32+lane][j]
__global__ __launch_bounds__(256) void cvt_w1(const float* __restrict__ w1,
                                              __bf16* __restrict__ w1b) {
  size_t s = (size_t)blockIdx.x * 256 + threadIdx.x;
  if (s >= (size_t)ND * NE * NF) return;
  int d = (int)(s >> 14);            // row (K dim), 16384 cols per row
  int c = (int)(s & 16383);
  int e = c >> 11, f = c & 2047;
  int ct = f >> 4, n = f & 15;
  int kk = d >> 5, r = d & 31;
  int lane = n + (r & 16);           // +16 for upper K half
  int j = r & 15;
  size_t dst = ((((size_t)e * 128 + ct) * 32 + kk) * 32 + lane) * 16 + j;
  w1b[dst] = (__bf16)w1[s];
}

// w2 [E*F][D] -> fragment-native: dest[((e*64+ct)*64+kk)*32+lane][j]
__global__ __launch_bounds__(256) void cvt_w2(const float* __restrict__ w2,
                                              __bf16* __restrict__ w2b) {
  size_t s = (size_t)blockIdx.x * 256 + threadIdx.x;
  if (s >= (size_t)NE * NF * ND) return;
  int row = (int)(s >> 10);
  int col = (int)(s & 1023);
  int e = row >> 11, k = row & 2047;
  int ct = col >> 4, n = col & 15;
  int kk = k >> 5, r = k & 31;
  int lane = n + (r & 16);
  int j = r & 15;
  size_t dst = ((((size_t)e * 64 + ct) * 64 + kk) * 32 + lane) * 16 + j;
  w2b[dst] = (__bf16)w2[s];
}

// ---------------- router: one wave per token ----------------
__global__ __launch_bounds__(256) void moe_router(const float* __restrict__ x,
                                                  const float* __restrict__ wr,
                                                  WS* __restrict__ ws) {
  __shared__ float pblk[8];
  __shared__ int   cblk[8];
  __shared__ float zblk;
  int tid = threadIdx.x;
  if (tid < 8) { pblk[tid] = 0.0f; cblk[tid] = 0; }
  if (tid == 8) zblk = 0.0f;
  __syncthreads();

  int wave = tid >> 5, lane = tid & 31;
  int t = blockIdx.x * 8 + wave;
  const float* xr = x + (size_t)t * ND;

  float acc[8];
#pragma unroll
  for (int e = 0; e < 8; ++e) acc[e] = 0.0f;
  for (int d = lane; d < ND; d += 32) {
    float xv = xr[d];
    const float* w = wr + (size_t)d * NE;
#pragma unroll
    for (int e = 0; e < 8; ++e) acc[e] += xv * w[e];
  }
  for (int off = 16; off > 0; off >>= 1) {
#pragma unroll
    for (int e = 0; e < 8; ++e) acc[e] += __shfl_xor(acc[e], off, 32);
  }

  if (lane == 0) {
    float m = acc[0];
#pragma unroll
    for (int e = 1; e < 8; ++e) m = fmaxf(m, acc[e]);
    float p[8]; float s = 0.0f;
#pragma unroll
    for (int e = 0; e < 8; ++e) { p[e] = expf(acc[e] - m); s += p[e]; }
    float inv = 1.0f / s;
#pragma unroll
    for (int e = 0; e < 8; ++e) p[e] *= inv;
    float lse = logf(s) + m;

    int i0 = 0;
#pragma unroll
    for (int e = 1; e < 8; ++e) if (p[e] > p[i0]) i0 = e;
    int i1 = (i0 == 0) ? 1 : 0;
#pragma unroll
    for (int e = 0; e < 8; ++e) if (e != i0 && p[e] > p[i1]) i1 = e;
    float sw = p[i0] + p[i1];
    ws->topi[t * 2 + 0] = i0;  ws->topw[t * 2 + 0] = p[i0] / sw;
    ws->topi[t * 2 + 1] = i1;  ws->topw[t * 2 + 1] = p[i1] / sw;

    atomicAdd(&cblk[i0], 1);
    atomicAdd(&cblk[i1], 1);
#pragma unroll
    for (int e = 0; e < 8; ++e) atomicAdd(&pblk[e], p[e]);
    atomicAdd(&zblk, lse * lse);
  }
  __syncthreads();
  if (tid < 8) {
    atomicAdd(&ws->counts[tid], cblk[tid]);
    atomicAdd(&ws->p_sum[tid], pblk[tid]);
  }
  if (tid == 8) atomicAdd(&ws->z_acc, zblk);
}

// ---------------- tiny prefix scan over 8 experts ----------------
__global__ void moe_scan(WS* ws) {
  if (threadIdx.x == 0 && blockIdx.x == 0) {
    int o = 0;
#pragma unroll
    for (int e = 0; e < 8; ++e) { ws->offsets[e] = o; o += ws->counts[e]; }
    ws->offsets[8] = o;
  }
}

// ---------------- scatter tokens into per-expert lists ----------------
__global__ __launch_bounds__(256) void moe_scatter(WS* __restrict__ ws) {
  __shared__ int lcount[8], lbase[8];
  int tid = threadIdx.x;
  if (tid < 8) lcount[tid] = 0;
  __syncthreads();
  int idx = blockIdx.x * 256 + tid;           // grid sized to exactly NT*NK
  int e = ws->topi[idx];
  int lpos = atomicAdd(&lcount[e], 1);
  __syncthreads();
  if (tid < 8) lbase[tid] = atomicAdd(&ws->cursors[tid], lcount[tid]);
  __syncthreads();
  int slot = ws->offsets[e] + lbase[e] + lpos;
  ws->row_token[slot] = idx >> 1;
  ws->row_wt[slot]    = ws->topw[idx];
}

// ---------------- fused expert FFN ----------------
// grid=(NT/16, NE); 256 threads = 8 waves; 16-token tile per block.
// As: 16 rows, TDM-padded stride 2112B (32B pad after each 1024B -> bank spread)
// Hs: 16 rows x 2056 bf16 (4112B stride)
#define AS_ROW_B  2112
#define AS_BYTES  (16 * AS_ROW_B)              // 33792
#define HS_STRIDE 2056
#define HS_BYTES  (16 * HS_STRIDE * 2)         // 65792
#define FFN_LDS   (AS_BYTES + HS_BYTES + 64 + 64)

__global__ __launch_bounds__(256) void moe_ffn(const __bf16* __restrict__ xb,
                                               const __bf16* __restrict__ w1b,
                                               const __bf16* __restrict__ w2b,
                                               float* __restrict__ out,
                                               const WS* __restrict__ ws) {
  int e = blockIdx.y;
  int tile = blockIdx.x;
  int cnt = ws->counts[e];
  if (tile * 16 >= cnt) return;               // block-uniform early exit
  int base = ws->offsets[e];

  extern __shared__ char smem[];
  char*   Asb = smem;
  __bf16* Hs  = (__bf16*)(smem + AS_BYTES);
  float*  lw  = (float*)(smem + AS_BYTES + HS_BYTES);
  int*    tok = (int*)(smem + AS_BYTES + HS_BYTES + 64);

  int tid = threadIdx.x;
  if (tid < 16) {
    int r = tile * 16 + tid;
    int cr = (r < cnt) ? r : (cnt - 1);       // pad rows duplicate last valid token
    int slot = base + cr;
    tok[tid] = ws->row_token[slot];
    lw[tid]  = (r < cnt) ? ws->row_wt[slot] : 0.0f;
  }
  __syncthreads();

#if USE_TDM
  // ---- TDM gather: 16 rows of xb (2KB each) -> LDS with pad_enable ----
  if (tid < 32) {
    unsigned long long ga = (unsigned long long)(size_t)xb;
    unsigned lds = (unsigned)(size_t)Asb;     // low 32 bits of LDS aperture = offset
    u32x4 g0 = { 1u | (1u << 31),             // count=1, gather_mode=1, 16-bit idx
                 lds,
                 (unsigned)ga,
                 (unsigned)((ga >> 32) & 0x1FFFFFFu) | (2u << 30) }; // type=2
    i32x8 g1 = { (int)((1u << 16) | (1u << 20) | (7u << 22) | (7u << 25)),
                 //  data_size=2B      pad_en    interval=256dw  amount=8dw
                 (int)((1024u & 0xffffu) << 16),           // tensor_dim0 lo
                 (int)(((1024u >> 16) & 0xffffu) | ((8192u & 0xffffu) << 16)),
                 (int)(((8192u >> 16) & 0xffffu) | (1024u << 16)),  // tile_dim0=1024
                 (int)16,                                  // tile_dim1 = 16 indices
                 (int)1024,                                // tensor_dim0_stride lo
                 0, 0 };
    int idx0 = (tok[0]  & 0xffff) | (tok[1]  << 16);
    int idx1 = (tok[2]  & 0xffff) | (tok[3]  << 16);
    int idx2 = (tok[4]  & 0xffff) | (tok[5]  << 16);
    int idx3 = (tok[6]  & 0xffff) | (tok[7]  << 16);
    int idx4 = (tok[8]  & 0xffff) | (tok[9]  << 16);
    int idx5 = (tok[10] & 0xffff) | (tok[11] << 16);
    int idx6 = (tok[12] & 0xffff) | (tok[13] << 16);
    int idx7 = (tok[14] & 0xffff) | (tok[15] << 16);
    i32x4 g2 = { idx0, idx1, idx2, idx3 };
    i32x4 g3 = { idx4, idx5, idx6, idx7 };
    __builtin_amdgcn_tensor_load_to_lds(g0, g1, g2, g3, 0);
    __builtin_amdgcn_s_wait_tensorcnt(0);
  }
#else
  { // manual staging with same padded layout: 16B chunk cc of row
    int row = tid >> 4, c0 = tid & 15;
    const u32x4* src = (const u32x4*)(xb + (size_t)tok[row] * ND);
    for (int cc = c0; cc < 128; cc += 16) {
      u32x4 v = src[cc];
      *(u32x4*)(Asb + row * AS_ROW_B + cc * 16 + ((cc >= 64) ? 32 : 0)) = v;
    }
  }
#endif
  __syncthreads();

  int wave = tid >> 5, lane = tid & 31;
  int lr = lane & 15, hi = lane >> 4;

  // ---- stage 1: H(16x2048) = Xg(16x1024) @ W1_e ----
  for (int ct = wave; ct < NF / 16; ct += 8) {
    int col0 = ct * 16;
    const __bf16* b1 = w1b + (((size_t)e * 128 + ct) * 32) * 512 + (size_t)lane * 16;
    v8f acc = {0.f, 0.f, 0.f, 0.f, 0.f, 0.f, 0.f, 0.f};
    for (int kk = 0; kk < ND / 32; ++kk) {
      int kb8 = kk * 32 + hi * 16;
      const char* ap = Asb + lr * AS_ROW_B + kb8 * 2 + ((kb8 & 512) ? 32 : 0);
      Frag16 a, b;
      a.u[0] = *(const u32x4*)ap;            // K +0..7   (hi: +16..23)
      a.u[1] = *(const u32x4*)(ap + 16);     // K +8..15  (hi: +24..31)
      const u32x4* bp = (const u32x4*)(b1 + (size_t)kk * 512);
      b.u[0] = bp[0];
      b.u[1] = bp[1];
      acc = __builtin_amdgcn_wmma_f32_16x16x32_bf16(
          false, a.v, false, b.v, (short)0, acc, false, false);
    }
#pragma unroll
    for (int r = 0; r < 8; ++r)
      Hs[(hi * 8 + r) * HS_STRIDE + col0 + lr] = (__bf16)acc[r];
  }
  __syncthreads();

  // ---- stage 2: O(16x1024) = H @ W2_e, scale, accumulate ----
  float wrow[8]; int trow[8];
#pragma unroll
  for (int r = 0; r < 8; ++r) { wrow[r] = lw[hi * 8 + r]; trow[r] = tok[hi * 8 + r]; }

  for (int ct = wave; ct < ND / 16; ct += 8) {
    int gcol = ct * 16 + lr;
    const __bf16* b2 = w2b + (((size_t)e * 64 + ct) * 64) * 512 + (size_t)lane * 16;
    v8f acc = {0.f, 0.f, 0.f, 0.f, 0.f, 0.f, 0.f, 0.f};
    for (int kk = 0; kk < NF / 32; ++kk) {
      const __bf16* ap = Hs + lr * HS_STRIDE + kk * 32 + hi * 16;
      Frag16 a, b;
      a.u[0] = *(const u32x4*)ap;
      a.u[1] = *(const u32x4*)(ap + 8);
      const u32x4* bp = (const u32x4*)(b2 + (size_t)kk * 512);
      b.u[0] = bp[0];
      b.u[1] = bp[1];
      acc = __builtin_amdgcn_wmma_f32_16x16x32_bf16(
          false, a.v, false, b.v, (short)0, acc, false, false);
    }
#pragma unroll
    for (int r = 0; r < 8; ++r) {
      float w = wrow[r];
      if (w != 0.0f)
        atomicAdd(&out[(size_t)trow[r] * ND + gcol], w * acc[r]);
    }
  }
}

// ---------------- finalize aux losses ----------------
__global__ void moe_final(float* __restrict__ out, const WS* __restrict__ ws) {
  int tid = threadIdx.x;   // 32 threads
  float part = 0.0f;
  if (tid < 8) {
    float fi = (float)ws->counts[tid] / (float)(NT * NK);
    out[OUTN + 2 + tid] = fi;
    float pi = ws->p_sum[tid] / (float)NT;
    part = fi * pi;
  }
  for (int off = 16; off > 0; off >>= 1) part += __shfl_xor(part, off, 32);
  if (tid == 0) {
    out[OUTN + 0] = ws->z_acc / (float)NT;      // mean(lse^2)
    out[OUTN + 1] = (float)NE * part;           // E * dot(f_i, p_i)
  }
}

// ---------------- launch ----------------
extern "C" void kernel_launch(void* const* d_in, const int* in_sizes, int n_in,
                              void* d_out, int out_size, void* d_ws, size_t ws_size,
                              hipStream_t stream) {
  const float* x   = (const float*)d_in[0];
  const float* wrt = (const float*)d_in[1];
  const float* w1  = (const float*)d_in[2];
  const float* w2  = (const float*)d_in[3];
  float* out = (float*)d_out;
  WS* ws = (WS*)d_ws;
  __bf16* xb  = (__bf16*)((char*)d_ws + XB_OFF);
  __bf16* w1b = (__bf16*)((char*)d_ws + W1B_OFF);
  __bf16* w2b = (__bf16*)((char*)d_ws + W2B_OFF);

  size_t zn = OUTN + 10;
  moe_init<<<(unsigned)((zn + 255) / 256), 256, 0, stream>>>(out, ws);
  cvt_x<<<(unsigned)(((size_t)NT * ND + 255) / 256), 256, 0, stream>>>(x, xb);
  cvt_w1<<<(unsigned)(((size_t)ND * NE * NF + 255) / 256), 256, 0, stream>>>(w1, w1b);
  cvt_w2<<<(unsigned)(((size_t)NE * NF * ND + 255) / 256), 256, 0, stream>>>(w2, w2b);
  moe_router<<<NT / 8, 256, 0, stream>>>(x, wrt, ws);
  moe_scan<<<1, 1, 0, stream>>>(ws);
  moe_scatter<<<(NT * NK) / 256, 256, 0, stream>>>(ws);
  dim3 g(NT / 16, NE);
  moe_ffn<<<g, 256, FFN_LDS, stream>>>(xb, w1b, w2b, out, ws);
  moe_final<<<1, 32, 0, stream>>>(out, ws);
}
